// CausalSelfAttention_79276506349575
// MI455X (gfx1250) — compile-verified, matrix operand order checked by reading, not run on hardware
//
#include <hip/hip_runtime.h>

typedef __attribute__((ext_vector_type(16))) _Float16 v16h;
typedef __attribute__((ext_vector_type(8)))  float    v8f;
typedef __attribute__((ext_vector_type(4)))  _Float16 v4h;
typedef __attribute__((ext_vector_type(4)))  float    v4f;

// D = A(16x32 f16) x B(32x16 f16) + C(16x16 f32)
#define WMMA_F16(a, b, c) \
  __builtin_amdgcn_wmma_f32_16x16x32_f16(false, (a), false, (b), (short)0, (c), false, false)

// Per-lane K-index mapping for 16-bit A/B fragments (16x32):
// element e of lane -> k = (e&7) + 16*(e>=8) + 8*(lane>=16)
__device__ __forceinline__ int kmap(int e, int hi) {
  return (e & 7) + ((e >> 3) << 4) + (hi << 3);
}

// ---------------------------------------------------------------------------
// CDNA5 async global->LDS copy (GLOBAL_LOAD_ASYNC_TO_LDS_B128, ASYNCcnt),
// with a synchronous VGPR-roundtrip fallback if the builtin is unavailable.
// Builtin signature (from clang diagnostic): param0 = vector_size(16) int* in
// the global (__device__) address space, non-const; param1 = LDS pointer;
// then two integer (offset, cpol) args.
// ---------------------------------------------------------------------------
#if defined(__HIP_DEVICE_COMPILE__) && __has_builtin(__builtin_amdgcn_global_load_async_to_lds_b128)
#define HAS_ASYNC_LDS 1
#else
#define HAS_ASYNC_LDS 0
#endif
#if defined(__HIP_DEVICE_COMPILE__) && __has_builtin(__builtin_amdgcn_s_wait_asynccnt)
#define HAS_WAIT_ASYNC 1
#else
#define HAS_WAIT_ASYNC 0
#endif

typedef int v4i __attribute__((vector_size(16)));
typedef __attribute__((address_space(1))) v4i g_v4i;
typedef __attribute__((address_space(3))) v4i l_v4i;

__device__ __forceinline__ void cp16(void* lds_dst, const void* gsrc) {
#if HAS_ASYNC_LDS
  // Generic LDS pointer low 32 bits are the LDS offset (ISA aperture rule).
  g_v4i* g = (g_v4i*)(uintptr_t)gsrc;
  l_v4i* l = (l_v4i*)(unsigned)(uintptr_t)lds_dst;
  __builtin_amdgcn_global_load_async_to_lds_b128(g, l, 0, 0);
#else
  *(uint4*)lds_dst = *(const uint4*)gsrc;
#endif
}

__device__ __forceinline__ void cp_wait() {
#if HAS_ASYNC_LDS
#if HAS_WAIT_ASYNC
  __builtin_amdgcn_s_wait_asynccnt(0);
#else
  asm volatile("s_wait_asynccnt 0x0" ::: "memory");
#endif
#endif
}

// ---------------------------------------------------------------------------
// Elementwise f32 -> f16 convert (vectorized, grid-strided).
// ---------------------------------------------------------------------------
__global__ __launch_bounds__(256) void cvt_f16_kernel(
    const float* __restrict__ in, _Float16* __restrict__ out, int n4) {
  for (int i = blockIdx.x * blockDim.x + threadIdx.x; i < n4;
       i += gridDim.x * blockDim.x) {
    v4f v = *(const v4f*)(in + (size_t)i * 4);
    v4h h;
#pragma unroll
    for (int j = 0; j < 4; ++j) h[j] = (_Float16)v[j];
    *(v4h*)(out + (size_t)i * 4) = h;
  }
}

// f32 [R][Cc] -> f16 transposed [Cc][R]
__global__ __launch_bounds__(256) void cvt_f16_t_kernel(
    const float* __restrict__ in, _Float16* __restrict__ out, int R, int Cc) {
  const int total = R * Cc;
  for (int i = blockIdx.x * blockDim.x + threadIdx.x; i < total;
       i += gridDim.x * blockDim.x) {
    const int n = i / R, k = i - n * R;  // out[n][k] = in[k][n]
    out[i] = (_Float16)in[(size_t)k * Cc + n];
  }
}

// ---------------------------------------------------------------------------
// Pipelined f16 GEMM with bias: C[M,N] = A[M,K] * Bt[N,K]^T + bias[N]
// BM=128, BN=128, BK=32; 256 threads = 8 waves (2m x 4n), 4x2 WMMA per wave.
// Double-buffered LDS tiles filled with async global->LDS b128 copies.
// Both A and B LDS tiles are [row][32 k] so every fragment load is b128.
// TRANSC: store transposed, C[N,M] (used for V so attention can async-stage
// V^T tiles contiguously).
// ---------------------------------------------------------------------------
template <typename OT, bool TRANSC>
__global__ __launch_bounds__(256) void gemm_f16_kernel(
    const _Float16* __restrict__ A,   // [M][K] f16
    const _Float16* __restrict__ Bt,  // [N][K] f16 (pre-transposed weights)
    const float* __restrict__ bias, OT* __restrict__ Cmat,
    int M, int N, int K) {
  __shared__ __align__(16) _Float16 As[2][128 * 32];
  __shared__ __align__(16) _Float16 Bs[2][128 * 32];

  const int tid  = threadIdx.x;
  const int lane = tid & 31;
  const int wave = tid >> 5;
  const int hi   = lane >> 4;
  const int lm   = lane & 15;
  const int wm   = (wave >> 2) * 64;
  const int wn   = (wave & 3) * 32;
  const int m0   = blockIdx.y * 128;
  const int n0   = blockIdx.x * 128;

  auto stage = [&](int buf, int k0) {
    // 512 16B chunks each for A and B; 4 async copies per thread.
#pragma unroll
    for (int i = 0; i < 2; ++i) {
      const int c = i * 256 + tid;
      const int row = c >> 2, kc = (c & 3) * 8;
      cp16(&As[buf][row * 32 + kc], A + (size_t)(m0 + row) * K + k0 + kc);
    }
#pragma unroll
    for (int i = 0; i < 2; ++i) {
      const int c = i * 256 + tid;
      const int row = c >> 2, kc = (c & 3) * 8;
      cp16(&Bs[buf][row * 32 + kc], Bt + (size_t)(n0 + row) * K + k0 + kc);
    }
  };

  v8f acc[4][2];
#pragma unroll
  for (int mt = 0; mt < 4; ++mt)
#pragma unroll
    for (int nt = 0; nt < 2; ++nt) acc[mt][nt] = v8f{};

  const int nk = K / 32;
  stage(0, 0);
  int buf = 0;
  for (int kt = 0; kt < nk; ++kt) {
    cp_wait();
    __syncthreads();  // tile kt visible to all; all reads of buf^1 finished
    if (kt + 1 < nk) stage(buf ^ 1, (kt + 1) * 32);  // overlap DMA with math

    v16h af[4], bf[2];
#pragma unroll
    for (int mt = 0; mt < 4; ++mt) {
      const int m = wm + mt * 16 + lm;
#pragma unroll
      for (int e = 0; e < 16; ++e) af[mt][e] = As[buf][m * 32 + kmap(e, hi)];
    }
#pragma unroll
    for (int nt = 0; nt < 2; ++nt) {
      const int n = wn + nt * 16 + lm;
#pragma unroll
      for (int e = 0; e < 16; ++e) bf[nt][e] = Bs[buf][n * 32 + kmap(e, hi)];
    }
#pragma unroll
    for (int mt = 0; mt < 4; ++mt)
#pragma unroll
      for (int nt = 0; nt < 2; ++nt)
        acc[mt][nt] = WMMA_F16(af[mt], bf[nt], acc[mt][nt]);
    buf ^= 1;
  }

  // Epilogue: C layout -> lane holds column n=lm, rows m = r + 8*hi
#pragma unroll
  for (int mt = 0; mt < 4; ++mt)
#pragma unroll
    for (int nt = 0; nt < 2; ++nt) {
      const int gn = n0 + wn + nt * 16 + lm;
      const float bv = bias[gn];
#pragma unroll
      for (int r = 0; r < 8; ++r) {
        const int gm = m0 + wm + mt * 16 + r + hi * 8;
        const float v = acc[mt][nt][r] + bv;
        if (TRANSC)
          Cmat[(size_t)gn * M + gm] = (OT)v;
        else
          Cmat[(size_t)gm * N + gn] = (OT)v;
      }
    }
}

// ---------------------------------------------------------------------------
// Causal flash attention, head dim 64. 128 threads = 4 waves x 16 queries.
// Double-buffered 32-key K/V tiles staged with async global->LDS copies.
// S^T = K*Q^T so softmax stats are in-lane + one shfl_xor(16); P^T f32
// accumulator maps element-for-element to the f16 B-fragment of
// O^T += V^T * P^T. V comes pre-transposed ([C][B*T]) so its tiles stage
// contiguously and its A-fragments load as b128.
// ---------------------------------------------------------------------------
__global__ __launch_bounds__(128) void flash_attn_kernel(
    const _Float16* __restrict__ Q,   // [B*T][C]
    const _Float16* __restrict__ Kd,  // [B*T][C]
    const _Float16* __restrict__ Vt,  // [C][B*T] (transposed)
    _Float16* __restrict__ Y,         // [B*T][C]
    int T, int C, int H, int Mrows) {
  constexpr int HD = 64;
  __shared__ __align__(16) _Float16 Ks[2][32 * HD];  // [key][d]
  __shared__ __align__(16) _Float16 Vs[2][HD * 32];  // [d][key]

  const int tid  = threadIdx.x;
  const int lane = tid & 31;
  const int wave = tid >> 5;
  const int hi   = lane >> 4;
  const int lm   = lane & 15;

  const int qtiles = T / 64;
  const int qt = blockIdx.x % qtiles;
  const int h  = (blockIdx.x / qtiles) % H;
  const int b  = blockIdx.x / (qtiles * H);
  const int qb = qt * 64;
  const int qg = qb + wave * 16 + lm;  // this lane's query row
  const size_t bT = (size_t)b * T;
  const size_t headoff = (size_t)h * HD;

  auto stage = [&](int buf, int kbase) {
    // K tile: 32 rows x 8 chunks; V tile: 64 rows x 4 chunks. 4 copies/thread.
#pragma unroll
    for (int i = 0; i < 2; ++i) {
      const int c = i * 128 + tid;
      const int row = c >> 3, dc = (c & 7) * 8;
      cp16(&Ks[buf][row * HD + dc],
           Kd + (bT + kbase + row) * C + headoff + dc);
    }
#pragma unroll
    for (int i = 0; i < 2; ++i) {
      const int c = i * 128 + tid;
      const int d = c >> 2, kc = (c & 3) * 8;
      cp16(&Vs[buf][d * 32 + kc],
           Vt + (headoff + d) * Mrows + bT + kbase + kc);
    }
  };

  // B fragments of Q^T: B[k=d][n=q] = Q[q][d]; two K-chunks (d 0..31, 32..63)
  v16h bq[2];
#pragma unroll
  for (int dk = 0; dk < 2; ++dk)
#pragma unroll
    for (int e = 0; e < 16; ++e) {
      const int d = kmap(e, hi) + dk * 32;
      bq[dk][e] = Q[(bT + qg) * C + headoff + d];
    }

  v8f od[4];  // O^T accumulators: 4 d-tiles of 16
#pragma unroll
  for (int dt = 0; dt < 4; ++dt) od[dt] = v8f{};
  float mrun = -1e30f, lrun = 0.0f;
  const float scale = 0.125f;  // 1/sqrt(64)

  const int nkt = (qb + 64) / 32;  // causal: only tiles with keys <= qb+63
  stage(0, 0);
  int buf = 0;
  for (int kt = 0; kt < nkt; ++kt) {
    const int kbase = kt * 32;
    cp_wait();
    __syncthreads();
    if (kt + 1 < nkt) stage(buf ^ 1, (kt + 1) * 32);

    // S^T tiles: rows = keys (2 tiles of 16), cols = this wave's 16 queries
    v8f st[2];
#pragma unroll
    for (int mt = 0; mt < 2; ++mt) {
      st[mt] = v8f{};
#pragma unroll
      for (int dk = 0; dk < 2; ++dk) {
        v16h ak;
        const int key = mt * 16 + lm;
#pragma unroll
        for (int e = 0; e < 16; ++e)
          ak[e] = Ks[buf][key * HD + kmap(e, hi) + dk * 32];
        st[mt] = WMMA_F16(ak, bq[dk], st[mt]);
      }
    }

    // Online softmax across this tile's 32 keys (per query column).
    float sv[16];
    float smax = -1e30f;
#pragma unroll
    for (int t = 0; t < 2; ++t)
#pragma unroll
      for (int r = 0; r < 8; ++r) {
        const int kglob = kbase + t * 16 + r + hi * 8;
        float s = st[t][r] * scale;
        s = (kglob > qg) ? -1e30f : s;
        sv[t * 8 + r] = s;
        smax = fmaxf(smax, s);
      }
    smax = fmaxf(smax, __shfl_xor(smax, 16, 32));
    const float mnew  = fmaxf(mrun, smax);
    const float alpha = __expf(mrun - mnew);

    v16h pf;  // P^T f16 B-fragment: element e <-> key kmap(e,hi) == sv index e
    float psum = 0.0f;
#pragma unroll
    for (int e = 0; e < 16; ++e) {
      const float p = __expf(sv[e] - mnew);
      psum += p;
      pf[e] = (_Float16)p;
    }
    psum += __shfl_xor(psum, 16, 32);
    lrun = lrun * alpha + psum;
    mrun = mnew;

#pragma unroll
    for (int dt = 0; dt < 4; ++dt) {
#pragma unroll
      for (int r = 0; r < 8; ++r) od[dt][r] *= alpha;
      v16h av;  // A fragment of V^T: row d = dt*16+lm, k = key (contiguous)
#pragma unroll
      for (int e = 0; e < 16; ++e)
        av[e] = Vs[buf][(dt * 16 + lm) * 32 + kmap(e, hi)];
      od[dt] = WMMA_F16(av, pf, od[dt]);
    }
    buf ^= 1;
  }

  // Epilogue: O^T C-layout -> lane holds query column qg, rows d = r + 8*hi
  const float inv = 1.0f / lrun;
#pragma unroll
  for (int dt = 0; dt < 4; ++dt)
#pragma unroll
    for (int r = 0; r < 8; ++r) {
      const int d = dt * 16 + r + hi * 8;
      Y[(bT + qg) * C + headoff + d] = (_Float16)(od[dt][r] * inv);
    }
}

// ---------------------------------------------------------------------------
extern "C" void kernel_launch(void* const* d_in, const int* in_sizes, int n_in,
                              void* d_out, int out_size, void* d_ws, size_t ws_size,
                              hipStream_t stream) {
  constexpr int B = 4, T = 2048, C = 1024, H = 16;
  constexpr int M = B * T;

  const float* x  = (const float*)d_in[0];
  const float* Wq = (const float*)d_in[1];
  const float* bq = (const float*)d_in[2];
  const float* Wk = (const float*)d_in[3];
  const float* bk = (const float*)d_in[4];
  const float* Wv = (const float*)d_in[5];
  const float* bv = (const float*)d_in[6];
  const float* Wp = (const float*)d_in[7];
  const float* bp = (const float*)d_in[8];
  float* out = (float*)d_out;

  // Workspace (66 MB): q, k, v^T, shared x/y buffer, one reusable weight slot.
  _Float16* qh = (_Float16*)d_ws;
  _Float16* kh = qh + (size_t)M * C;
  _Float16* vt = kh + (size_t)M * C;
  _Float16* xy = vt + (size_t)M * C;  // x (f16) in phase 1, y in phase 2
  _Float16* wh = xy + (size_t)M * C;  // 2 MB transposed-weight slot (reused)

  dim3 ggrid(C / 128, M / 128);

  cvt_f16_kernel<<<2048, 256, 0, stream>>>(x, xy, M * C / 4);

  cvt_f16_t_kernel<<<1024, 256, 0, stream>>>(Wq, wh, C, C);
  gemm_f16_kernel<_Float16, false><<<ggrid, 256, 0, stream>>>(xy, wh, bq, qh, M, C, C);
  cvt_f16_t_kernel<<<1024, 256, 0, stream>>>(Wk, wh, C, C);
  gemm_f16_kernel<_Float16, false><<<ggrid, 256, 0, stream>>>(xy, wh, bk, kh, M, C, C);
  cvt_f16_t_kernel<<<1024, 256, 0, stream>>>(Wv, wh, C, C);
  gemm_f16_kernel<_Float16, true><<<ggrid, 256, 0, stream>>>(xy, wh, bv, vt, M, C, C);

  flash_attn_kernel<<<B * H * (T / 64), 128, 0, stream>>>(qh, kh, vt, xy, T, C, H, M);

  cvt_f16_t_kernel<<<1024, 256, 0, stream>>>(Wp, wh, C, C);
  gemm_f16_kernel<float, false><<<ggrid, 256, 0, stream>>>(xy, wh, bp, out, M, C, C);
}
